// GCNIIBackbone_77378130804856
// MI455X (gfx1250) — matrix operand order repaired and checked by reading, not compiled, not run on hardware
//
#include <hip/hip_runtime.h>
#include <hip/hip_bf16.h>
#include <math.h>

typedef __attribute__((ext_vector_type(16))) __bf16 v16bf;
typedef __attribute__((ext_vector_type(8)))  float  v8f;

#define N_NODES 100000
#define E_EDGES 1600000
#define DIN     256
#define DMID    128
#define NLAYERS 4
#define ALPHA_C 0.5f
#define EPS_C   1e-5f
#define ND      (N_NODES * DMID)

// ---------------------------------------------------------------------------
// WMMA fragment loaders (CDNA5 wave32 layouts, 16x16x32 bf16)
// ---------------------------------------------------------------------------

// A (16x32, MxK) from fp32 row-major [.. x ld], rows m0..m0+15, cols k0..k0+31,
// scaled by `scale`, converted to bf16.
// Layout: lane l (l<16): M=l, VGPR0-7 hold K=0..15;
//         lane l+16:     M=l, VGPR0-7 hold K=16..31.
__device__ __forceinline__ v16bf load_a_f32(const float* __restrict__ src, int ld,
                                            int m0, int k0, float scale, int lane) {
  const int m  = m0 + (lane & 15);
  const int kh = (lane >> 4) << 4;                 // 0 or 16
  const float* __restrict__ p = src + (size_t)m * (size_t)ld;
  v16bf a;
#pragma unroll
  for (int v = 0; v < 8; ++v) {
    const int kb = k0 + kh + 2 * v;
    a[2 * v]     = (__bf16)(p[kb]     * scale);
    a[2 * v + 1] = (__bf16)(p[kb + 1] * scale);
  }
  return a;
}

// B fragment from pre-swizzled weight buffer: wf[kt][nt][lane][16 elems],
// each lane's 16 bf16 contiguous -> one 32B vector load (2x b128).
__device__ __forceinline__ v16bf load_b_frag(const __bf16* __restrict__ wf,
                                             int kt, int nt, int lane) {
  return *(const v16bf*)(wf + ((((size_t)kt * 8 + nt) * 32 + lane) << 4));
}

// ---------------------------------------------------------------------------
// Weight prep: bf16 copies, pre-swizzled into WMMA B-fragment order.
// Fragment element (kt,nt,lane,e) <- src(K = kt*32 + (lane>>4)*16 + e,
//                                        N = nt*16 + (lane&15))
// ---------------------------------------------------------------------------
__global__ void k_prep_weights(const float* __restrict__ lin1_w,
                               const float* __restrict__ w1,
                               const float* __restrict__ w2,
                               __bf16* __restrict__ wtb,
                               __bf16* __restrict__ w1b,
                               __bf16* __restrict__ w2b) {
  const int i = blockIdx.x * blockDim.x + threadIdx.x;
  if (i < DIN * DMID) {
    // lin1: B = lin1_w.T  (src(K,N) = lin1_w[N*DIN + K]); kt in 0..7
    const int e = i & 15, lane = (i >> 4) & 31, nt = (i >> 9) & 7, kt = i >> 12;
    const int K = kt * 32 + ((lane >> 4) << 4) + e;
    const int n = nt * 16 + (lane & 15);
    wtb[i] = (__bf16)lin1_w[n * DIN + K];
  }
  const int j = i - DIN * DMID;
  if (j >= 0 && j < NLAYERS * DMID * DMID) {
    // layers: B = w[ly] row-major [K x N]; kt in 0..3
    const int e = j & 15, lane = (j >> 4) & 31, nt = (j >> 9) & 7;
    const int kt = (j >> 12) & 3, ly = j >> 14;
    const int K = kt * 32 + ((lane >> 4) << 4) + e;
    const int n = nt * 16 + (lane & 15);
    const size_t src = (size_t)ly * DMID * DMID + (size_t)K * DMID + n;
    w1b[j] = (__bf16)w1[src];
    w2b[j] = (__bf16)w2[src];
  }
}

// ---------------------------------------------------------------------------
// Degree / normalization
// ---------------------------------------------------------------------------
__global__ void k_deg_init(float* __restrict__ deg) {
  const int i = blockIdx.x * blockDim.x + threadIdx.x;
  if (i < N_NODES) deg[i] = 1.0f;                  // self-loop
}

__global__ void k_deg_edges(const int* __restrict__ col, float* __restrict__ deg) {
  const int e = blockIdx.x * blockDim.x + threadIdx.x;
  if (e < E_EDGES) atomicAdd(&deg[col[e]], 1.0f);
}

__global__ void k_rsqrt_inplace(float* __restrict__ deg) {
  const int i = blockIdx.x * blockDim.x + threadIdx.x;
  if (i < N_NODES) deg[i] = rsqrtf(deg[i]);        // now holds dis = deg^-1/2
}

// ---------------------------------------------------------------------------
// lin1: f = relu(x @ lin1_w.T + b), x0 = ALPHA * f
// One wave computes a 16x128 row-stripe (8 WMMA accumulators); A loaded once
// per k-step and reused across all 8 n-tiles.
// grid: ceil(N/64) blocks x 128 threads (4 waves x 16 rows).
// ---------------------------------------------------------------------------
__global__ __launch_bounds__(128) void k_lin1(const float* __restrict__ x,
                                              const __bf16* __restrict__ wtb,
                                              const float* __restrict__ bias,
                                              float* __restrict__ f,
                                              float* __restrict__ x0) {
  const int lane = threadIdx.x & 31;
  const int wv   = threadIdx.x >> 5;
  const int m0   = blockIdx.x * 64 + wv * 16;
  if (m0 >= N_NODES) return;                       // wave-uniform guard
  v8f acc[8] = {};
#pragma unroll
  for (int kt = 0; kt < DIN / 32; ++kt) {
    const v16bf a = load_a_f32(x, DIN, m0, kt * 32, 1.0f, lane);
#pragma unroll
    for (int nt = 0; nt < 8; ++nt) {
      const v16bf b = load_b_frag(wtb, kt, nt, lane);
      acc[nt] = __builtin_amdgcn_wmma_f32_16x16x32_bf16(false, a, false, b,
                                                        (short)0, acc[nt],
                                                        false, false);
    }
  }
  const int nl    = lane & 15;
  const int mbase = m0 + ((lane >> 4) << 3);
#pragma unroll
  for (int nt = 0; nt < 8; ++nt) {
    const int n = nt * 16 + nl;
    const float bv = bias[n];
#pragma unroll
    for (int r = 0; r < 8; ++r) {
      const size_t idx = (size_t)(mbase + r) * DMID + n;
      float v = acc[nt][r] + bv;
      v = v > 0.0f ? v : 0.0f;
      f[idx]  = v;
      x0[idx] = ALPHA_C * v;
    }
  }
}

// ---------------------------------------------------------------------------
// SpMM: h = segment_sum(f[row] * enorm, col)  (self-loops folded into init)
// ---------------------------------------------------------------------------
__global__ void k_spmm_self(const float* __restrict__ f,
                            const float* __restrict__ dis,
                            float* __restrict__ h) {
  const int i = blockIdx.x * blockDim.x + threadIdx.x;
  if (i >= ND) return;
  const float d = dis[i >> 7];                     // node = i / 128
  h[i] = f[i] * d * d;
}

__global__ __launch_bounds__(256) void k_spmm_edges(const int* __restrict__ row,
                                                    const int* __restrict__ col,
                                                    const float* __restrict__ dis,
                                                    const float* __restrict__ f,
                                                    float* __restrict__ h) {
  const int gt   = blockIdx.x * blockDim.x + threadIdx.x;
  const int e    = gt >> 5;                        // one wave32 per edge
  const int lane = gt & 31;
  if (e >= E_EDGES) return;
  const int r = row[e];
  const int c = col[e];
  const float w = dis[r] * dis[c];
  const float4 fv = *(const float4*)(f + (size_t)r * DMID + lane * 4);
  float* hp = h + (size_t)c * DMID + lane * 4;
  atomicAdd(hp + 0, fv.x * w);
  atomicAdd(hp + 1, fv.y * w);
  atomicAdd(hp + 2, fv.z * w);
  atomicAdd(hp + 3, fv.w * w);
}

__global__ void k_zero_stats(float* __restrict__ stats) {
  stats[0] = 0.0f;
  stats[1] = 0.0f;
}

// ---------------------------------------------------------------------------
// combine: out = (1-b)*hs + b*(hs@w1) + (1-b)*x0 + b*(x0@w2), hs = 0.5*h
// 16x128 row-stripe per wave; also accumulates global sum / sumsq for LN.
// ---------------------------------------------------------------------------
__global__ __launch_bounds__(128) void k_combine(const float* __restrict__ h,
                                                 const float* __restrict__ x0,
                                                 const __bf16* __restrict__ w1b,
                                                 const __bf16* __restrict__ w2b,
                                                 float* __restrict__ out,
                                                 float* __restrict__ stats,
                                                 float beta) {
  const int lane = threadIdx.x & 31;
  const int wv   = threadIdx.x >> 5;
  const int m0   = blockIdx.x * 64 + wv * 16;
  __shared__ float ssum[4], ssq[4];
  float lsum = 0.0f, lsq = 0.0f;
  if (m0 < N_NODES) {                              // wave-uniform guard
    v8f acc[8] = {};
#pragma unroll
    for (int kt = 0; kt < DMID / 32; ++kt) {
      const v16bf a1 = load_a_f32(h, DMID, m0, kt * 32, 1.0f - ALPHA_C, lane);
      const v16bf a2 = load_a_f32(x0, DMID, m0, kt * 32, 1.0f, lane);
#pragma unroll
      for (int nt = 0; nt < 8; ++nt) {
        const v16bf b1 = load_b_frag(w1b, kt, nt, lane);
        acc[nt] = __builtin_amdgcn_wmma_f32_16x16x32_bf16(false, a1, false, b1,
                                                          (short)0, acc[nt],
                                                          false, false);
        const v16bf b2 = load_b_frag(w2b, kt, nt, lane);
        acc[nt] = __builtin_amdgcn_wmma_f32_16x16x32_bf16(false, a2, false, b2,
                                                          (short)0, acc[nt],
                                                          false, false);
      }
    }
    const int nl    = lane & 15;
    const int mbase = m0 + ((lane >> 4) << 3);
#pragma unroll
    for (int nt = 0; nt < 8; ++nt) {
      const int n = nt * 16 + nl;
#pragma unroll
      for (int r = 0; r < 8; ++r) {
        const size_t idx = (size_t)(mbase + r) * DMID + n;
        const float hs = (1.0f - ALPHA_C) * h[idx];
        const float o  = beta * acc[nt][r] + (1.0f - beta) * (hs + x0[idx]);
        out[idx] = o;
        lsum += o;
        lsq  += o * o;
      }
    }
  }
#pragma unroll
  for (int off = 16; off; off >>= 1) {
    lsum += __shfl_xor(lsum, off, 32);
    lsq  += __shfl_xor(lsq, off, 32);
  }
  if (lane == 0) { ssum[wv] = lsum; ssq[wv] = lsq; }
  __syncthreads();
  if (threadIdx.x == 0) {
    atomicAdd(&stats[0], ssum[0] + ssum[1] + ssum[2] + ssum[3]);
    atomicAdd(&stats[1], ssq[0] + ssq[1] + ssq[2] + ssq[3]);
  }
}

// ---------------------------------------------------------------------------
// graph LayerNorm (global mean/std over all N*D) + affine + ReLU -> f
// ---------------------------------------------------------------------------
__global__ void k_ln_relu(const float* __restrict__ out,
                          const float* __restrict__ stats,
                          const float* __restrict__ nw,
                          const float* __restrict__ nb,
                          float* __restrict__ f) {
  const int i = blockIdx.x * blockDim.x + threadIdx.x;
  if (i >= ND) return;
  const float inv = 1.0f / (float)ND;
  const float m   = stats[0] * inv;
  float var = stats[1] * inv - m * m;
  var = var > 0.0f ? var : 0.0f;
  const float rs = 1.0f / (sqrtf(var) + EPS_C);
  const int n = i & (DMID - 1);
  float o = (out[i] - m) * rs * nw[n] + nb[n];
  f[i] = o > 0.0f ? o : 0.0f;
}

// ---------------------------------------------------------------------------
// Host launcher
// ---------------------------------------------------------------------------
extern "C" void kernel_launch(void* const* d_in, const int* in_sizes, int n_in,
                              void* d_out, int out_size, void* d_ws, size_t ws_size,
                              hipStream_t stream) {
  (void)in_sizes; (void)n_in; (void)out_size; (void)ws_size;

  const float* x      = (const float*)d_in[0];     // [N, DIN]
  const int*   ei     = (const int*)d_in[1];       // [2, E]
  const float* lin1_w = (const float*)d_in[2];     // [DMID, DIN]
  const float* lin1_b = (const float*)d_in[3];     // [DMID]
  const float* w1     = (const float*)d_in[4];     // [L, DMID, DMID]
  const float* w2     = (const float*)d_in[5];     // [L, DMID, DMID]
  const float* norm_w = (const float*)d_in[6];     // [DMID]
  const float* norm_b = (const float*)d_in[7];     // [DMID]

  const int* row = ei;                             // sources
  const int* col = ei + E_EDGES;                   // aggregation targets

  float* f = (float*)d_out;                        // activations live in d_out

  // Workspace carve-up (256B-aligned slabs)
  char* ws = (char*)d_ws;
  size_t off = 0;
  auto carve = [&](size_t bytes) -> char* {
    char* p = ws + off;
    off += (bytes + 255) & ~(size_t)255;
    return p;
  };
  float*  stats = (float*)carve(2 * sizeof(float));
  float*  dis   = (float*)carve((size_t)N_NODES * sizeof(float)); // deg -> dis
  float*  x0    = (float*)carve((size_t)ND * sizeof(float));
  float*  h     = (float*)carve((size_t)ND * sizeof(float));
  float*  outb  = (float*)carve((size_t)ND * sizeof(float));
  __bf16* wtb   = (__bf16*)carve((size_t)DIN * DMID * sizeof(__bf16));
  __bf16* w1b   = (__bf16*)carve((size_t)NLAYERS * DMID * DMID * sizeof(__bf16));
  __bf16* w2b   = (__bf16*)carve((size_t)NLAYERS * DMID * DMID * sizeof(__bf16));

  const int prep_total = DIN * DMID + NLAYERS * DMID * DMID;
  k_prep_weights<<<(prep_total + 255) / 256, 256, 0, stream>>>(lin1_w, w1, w2,
                                                               wtb, w1b, w2b);

  k_deg_init<<<(N_NODES + 255) / 256, 256, 0, stream>>>(dis);
  k_deg_edges<<<(E_EDGES + 255) / 256, 256, 0, stream>>>(col, dis);
  k_rsqrt_inplace<<<(N_NODES + 255) / 256, 256, 0, stream>>>(dis);

  const int ggrid = (N_NODES + 63) / 64;           // 4 waves x 16 rows / block
  k_lin1<<<ggrid, 128, 0, stream>>>(x, wtb, lin1_b, f, x0);

  const int nd_blocks   = (ND + 255) / 256;
  const int edge_blocks = (E_EDGES * 32 + 255) / 256;

  for (int i = 0; i < NLAYERS; ++i) {
    const float beta = logf(1.0f / (float)(i + 1) + 1.0f);  // THETA = 1
    k_spmm_self<<<nd_blocks, 256, 0, stream>>>(f, dis, h);
    k_spmm_edges<<<edge_blocks, 256, 0, stream>>>(row, col, dis, f, h);
    k_zero_stats<<<1, 1, 0, stream>>>(stats);
    k_combine<<<ggrid, 128, 0, stream>>>(h, x0,
                                         w1b + (size_t)i * DMID * DMID,
                                         w2b + (size_t)i * DMID * DMID,
                                         outb, stats, beta);
    k_ln_relu<<<nd_blocks, 256, 0, stream>>>(outb, stats, norm_w, norm_b, f);
  }
  // final relu(f) is idempotent (f already ReLU'd); d_out == f holds the result
}
